// InverseCrossAttentionMH_80547816669196
// MI455X (gfx1250) — matrix operand order repaired and checked by reading, not compiled, not run on hardware
//
#include <hip/hip_runtime.h>
#include <hip/hip_bf16.h>

// ---- WMMA bf16 types (gfx1250) ----
typedef __attribute__((ext_vector_type(16))) __bf16 v16bf;
typedef __attribute__((ext_vector_type(8)))  float  v8f;

struct __align__(16) U4  { unsigned int x, y, z, w; };
struct __align__(16) F4  { float x, y, z, w; };
struct __align__(8)  BH4 { __bf16 h[4]; };
union  Frag { v16bf v; U4 q[2]; };
union  U4H  { U4 u; __bf16 h[8]; };

#define DMODEL 1024
#define HEADS  16
#define DH     64
#define NQ     128
#define LK     4096
#define BATCH  8

static __device__ inline v8f zero8() {
  v8f z = {0.f,0.f,0.f,0.f,0.f,0.f,0.f,0.f};
  return z;
}
static __device__ inline v16bf ld_frag(const __bf16* p0, const __bf16* p1) {
  Frag f;
  f.q[0] = *(const U4*)p0;
  f.q[1] = *(const U4*)p1;
  return f.v;
}
static __device__ inline v8f wmma_bf16(v16bf a, v16bf b, v8f c) {
  return __builtin_amdgcn_wmma_f32_16x16x32_bf16(false, a, false, b, (short)0, c, false, false);
}

// Async per-lane 16B copy global -> LDS (gfx1250 GLOBAL_LOAD_ASYNC_TO_LDS_B128).
// lds generic pointer's low 32 bits are the LDS offset the DS unit consumes.
static __device__ inline void async_copy_b128(void* lds_dst, const void* gsrc) {
  unsigned dst = (unsigned)(unsigned long long)(uintptr_t)lds_dst;
  unsigned long long src = (unsigned long long)(uintptr_t)gsrc;
  asm volatile("global_load_async_to_lds_b128 %0, %1, off"
               :: "v"(dst), "v"(src) : "memory");
}
static __device__ inline void wait_asynccnt0() {
  asm volatile("s_wait_asynccnt 0" ::: "memory");
}

// =====================================================================
// Kernel 1: y = x @ W^T  (fp32 in, bf16 out, head-permuted store)
// grid = (M/128, 1024/64), block = 256 (8 waves)
// out layout: [b][h][row_in_batch][dh], b = m / RPB  (RPB = power of 2)
// =====================================================================
template <int RPB>
__global__ __launch_bounds__(256)
void proj_gemm_bf16(const float* __restrict__ A, const float* __restrict__ W,
                    __bf16* __restrict__ out) {
  __shared__ __bf16 As[128 * 32];
  __shared__ __bf16 Ws[64 * 32];

  const int tid  = threadIdx.x;
  const int w    = tid >> 5;
  const int lane = tid & 31;
  const int hi   = lane >> 4;
  const int li   = lane & 15;
  const int m0   = blockIdx.x * 128;
  const int n0   = blockIdx.y * 64;

  v8f acc[4];
  for (int i = 0; i < 4; ++i) acc[i] = zero8();

  for (int kt = 0; kt < DMODEL / 32; ++kt) {
    const int k0 = kt * 32;
    // A tile: 128x32 fp32 -> bf16 LDS (coalesced 16B loads)
    for (int i = 0; i < 4; ++i) {
      int idx = tid + i * 256;
      int row = idx >> 3, cg = idx & 7;
      F4 f = *(const F4*)(A + (size_t)(m0 + row) * DMODEL + k0 + cg * 4);
      BH4 bh; bh.h[0]=(__bf16)f.x; bh.h[1]=(__bf16)f.y; bh.h[2]=(__bf16)f.z; bh.h[3]=(__bf16)f.w;
      *(BH4*)(As + row * 32 + cg * 4) = bh;
    }
    // W tile: 64x32 fp32 -> bf16 LDS
    for (int i = 0; i < 2; ++i) {
      int idx = tid + i * 256;
      int row = idx >> 3, cg = idx & 7;
      F4 f = *(const F4*)(W + (size_t)(n0 + row) * DMODEL + k0 + cg * 4);
      BH4 bh; bh.h[0]=(__bf16)f.x; bh.h[1]=(__bf16)f.y; bh.h[2]=(__bf16)f.z; bh.h[3]=(__bf16)f.w;
      *(BH4*)(Ws + row * 32 + cg * 4) = bh;
    }
    __syncthreads();

    const int arow = w * 16 + li;
    const __bf16* ap = As + arow * 32 + hi * 8;
    v16bf a = ld_frag(ap, ap + 16);
    for (int nt = 0; nt < 4; ++nt) {
      const __bf16* bp = Ws + (nt * 16 + li) * 32 + hi * 8;
      v16bf b = ld_frag(bp, bp + 16);
      acc[nt] = wmma_bf16(a, b, acc[nt]);
    }
    __syncthreads();
  }

  // Store with head permute (RPB is a compile-time power of two -> shifts)
  for (int nt = 0; nt < 4; ++nt)
    for (int r = 0; r < 8; ++r) {
      int m  = m0 + w * 16 + r + hi * 8;
      int j  = n0 + nt * 16 + li;
      int b  = m / RPB, rr = m & (RPB - 1);
      int h  = j >> 6,  dh = j & 63;
      out[(((size_t)b * HEADS + h) * RPB + rr) * DH + dh] = (__bf16)acc[nt][r];
    }
}

// =====================================================================
// Kernel 2: inverted attention, one block per (b,h), streaming over L
// softmax over N (column-wise, local to chunk), L1 renorm over L (streamed)
// =====================================================================
__global__ __launch_bounds__(256)
void inv_attn(const __bf16* __restrict__ qh, const __bf16* __restrict__ kh,
              const __bf16* __restrict__ vh, __bf16* __restrict__ ao) {
  __shared__ __bf16 khs[64 * 64];        // kh chunk, row-major [l][c]
  __shared__ __bf16 vts[64 * 64];        // vh chunk, transposed [c][l]
  __shared__ __bf16 Ab[8 * 16 * 64];     // per-wave normalized-A strips
  __shared__ float  red[2048];           // reduction scratch (16x64 / rowsum)
  __shared__ float  cmax[64];
  __shared__ float  csum[64];
  __shared__ float  rinv[128];

  const int tid  = threadIdx.x;
  const int w    = tid >> 5;
  const int lane = tid & 31;
  const int hi   = lane >> 4;
  const int li   = lane & 15;
  const int bh   = blockIdx.x;           // b*16 + h
  const int b    = bh >> 4, h = bh & 15;
  const float scale = 0.03125f;          // 1024^-0.5

  const size_t qbase = (size_t)bh * NQ * DH;
  const size_t kbase = (size_t)bh * LK * DH;

  // per-wave q A-fragments (rows w*16..+15, K = 0..63)
  v16bf aq[2];
  {
    const __bf16* qp = qh + qbase + (size_t)(w * 16 + li) * DH;
    aq[0] = ld_frag(qp + hi * 8,      qp + hi * 8 + 16);
    aq[1] = ld_frag(qp + 32 + hi * 8, qp + 32 + hi * 8 + 16);
  }

  v8f accO[4];
  for (int i = 0; i < 4; ++i) accO[i] = zero8();
  float rs[8] = {0.f,0.f,0.f,0.f,0.f,0.f,0.f,0.f};

  for (int ic = 0; ic < LK / 64; ++ic) {
    const int l0 = ic * 64;

    // prefetch next chunk into cache while this one computes
    if (ic + 1 < LK / 64) {
      const int pr = tid >> 2;  // 0..63
      __builtin_prefetch(kh + kbase + (size_t)(l0 + 64 + pr) * DH, 0, 1);
      __builtin_prefetch(vh + kbase + (size_t)(l0 + 64 + pr) * DH, 0, 1);
    }

    // kh chunk: pure copy -> async DMA directly into LDS (ASYNCcnt-tracked)
    // vh chunk: loaded through VGPRs and stored transposed
    for (int i = 0; i < 2; ++i) {
      int idx = tid + i * 256;
      int row = idx >> 3, grp = idx & 7;
      async_copy_b128(khs + row * 64 + grp * 8,
                      kh + kbase + (size_t)(l0 + row) * DH + grp * 8);
      U4H t;
      t.u = *(const U4*)(vh + kbase + (size_t)(l0 + row) * DH + grp * 8);
      for (int j2 = 0; j2 < 8; ++j2)
        vts[(grp * 8 + j2) * 64 + row] = t.h[j2];
    }
    wait_asynccnt0();
    __syncthreads();

    // S strip: 16 rows x 64 cols, S = scale * qh @ kh^T
    v8f s[4];
    for (int nt = 0; nt < 4; ++nt) s[nt] = zero8();
    for (int ks = 0; ks < 2; ++ks) {
      for (int nt = 0; nt < 4; ++nt) {
        const __bf16* bp = khs + (nt * 16 + li) * 64 + hi * 8 + ks * 32;
        v16bf bf = ld_frag(bp, bp + 16);
        s[nt] = wmma_bf16(aq[ks], bf, s[nt]);
      }
    }
    for (int nt = 0; nt < 4; ++nt)
      for (int r = 0; r < 8; ++r) s[nt][r] *= scale;

    // column-wise (over N) max: 16 partials per column
    const int p = w * 2 + hi;
    for (int nt = 0; nt < 4; ++nt) {
      float m = s[nt][0];
      for (int r = 1; r < 8; ++r) m = fmaxf(m, s[nt][r]);
      red[p * 64 + nt * 16 + li] = m;
    }
    __syncthreads();
    if (tid < 64) {
      float m = red[tid];
      for (int pp = 1; pp < 16; ++pp) m = fmaxf(m, red[pp * 64 + tid]);
      cmax[tid] = m;
    }
    __syncthreads();
    // exp + column sum
    for (int nt = 0; nt < 4; ++nt) {
      float cm = cmax[nt * 16 + li];
      float ps = 0.f;
      for (int r = 0; r < 8; ++r) {
        float e = __expf(s[nt][r] - cm);
        s[nt][r] = e;
        ps += e;
      }
      red[p * 64 + nt * 16 + li] = ps;
    }
    __syncthreads();
    if (tid < 64) {
      float sm = 0.f;
      for (int pp = 0; pp < 16; ++pp) sm += red[pp * 64 + tid];
      csum[tid] = sm;
    }
    __syncthreads();
    // normalize, accumulate row-sums, write bf16 A strip (C->A relayout via LDS)
    __bf16* abw = Ab + w * (16 * 64);
    for (int nt = 0; nt < 4; ++nt) {
      float inv = 1.f / csum[nt * 16 + li];
      for (int r = 0; r < 8; ++r) {
        float a = s[nt][r] * inv;
        rs[r] += a;
        abw[(r + hi * 8) * 64 + nt * 16 + li] = (__bf16)a;
      }
    }
    __syncthreads();

    // out += A_strip(16x64) @ vh_chunk(64x64)
    for (int kl = 0; kl < 2; ++kl) {
      const __bf16* ap = abw + li * 64 + hi * 8 + kl * 32;
      v16bf a = ld_frag(ap, ap + 16);
      for (int ct = 0; ct < 4; ++ct) {
        const __bf16* bp = vts + (ct * 16 + li) * 64 + hi * 8 + kl * 32;
        v16bf bf = ld_frag(bp, bp + 16);
        accO[ct] = wmma_bf16(a, bf, accO[ct]);
      }
    }
    __syncthreads();
  }

  // reduce row-sums (L1 renorm) across the 16 lanes sharing each row
  for (int r = 0; r < 8; ++r) red[w * 256 + lane * 8 + r] = rs[r];
  __syncthreads();
  if (tid < 128) {
    int n = tid, w2 = n >> 4, rl = n & 15, hi2 = rl >> 3, r2 = rl & 7;
    float sm = 0.f;
    for (int l2 = 0; l2 < 16; ++l2) sm += red[w2 * 256 + (hi2 * 16 + l2) * 8 + r2];
    rinv[n] = 1.f / sm;
  }
  __syncthreads();

  // store to [b][n][h*64+dh] bf16
  for (int ct = 0; ct < 4; ++ct)
    for (int r = 0; r < 8; ++r) {
      int n  = w * 16 + r + hi * 8;
      int dh = ct * 16 + li;
      float vo = accO[ct][r] * rinv[n];
      ao[((size_t)(b * NQ + n)) * DMODEL + h * DH + dh] = (__bf16)vo;
    }
}

// =====================================================================
// Kernel 3: out = Ao(bf16, 1024x1024) @ Wo^T (fp32) -> fp32
// grid = (8, 16), block = 256
// =====================================================================
__global__ __launch_bounds__(256)
void out_gemm_f32(const __bf16* __restrict__ A, const float* __restrict__ W,
                  float* __restrict__ out) {
  __shared__ __bf16 As[128 * 32];
  __shared__ __bf16 Ws[64 * 32];

  const int tid  = threadIdx.x;
  const int w    = tid >> 5;
  const int lane = tid & 31;
  const int hi   = lane >> 4;
  const int li   = lane & 15;
  const int m0   = blockIdx.x * 128;
  const int n0   = blockIdx.y * 64;

  v8f acc[4];
  for (int i = 0; i < 4; ++i) acc[i] = zero8();

  for (int kt = 0; kt < DMODEL / 32; ++kt) {
    const int k0 = kt * 32;
    for (int i = 0; i < 2; ++i) {
      int idx = tid + i * 256;
      int row = idx >> 2, grp = idx & 3;
      *(U4*)(As + row * 32 + grp * 8) =
          *(const U4*)(A + (size_t)(m0 + row) * DMODEL + k0 + grp * 8);
    }
    for (int i = 0; i < 2; ++i) {
      int idx = tid + i * 256;
      int row = idx >> 3, cg = idx & 7;
      F4 f = *(const F4*)(W + (size_t)(n0 + row) * DMODEL + k0 + cg * 4);
      BH4 bh; bh.h[0]=(__bf16)f.x; bh.h[1]=(__bf16)f.y; bh.h[2]=(__bf16)f.z; bh.h[3]=(__bf16)f.w;
      *(BH4*)(Ws + row * 32 + cg * 4) = bh;
    }
    __syncthreads();

    const __bf16* ap = As + (w * 16 + li) * 32 + hi * 8;
    v16bf a = ld_frag(ap, ap + 16);
    for (int nt = 0; nt < 4; ++nt) {
      const __bf16* bp = Ws + (nt * 16 + li) * 32 + hi * 8;
      v16bf b = ld_frag(bp, bp + 16);
      acc[nt] = wmma_bf16(a, b, acc[nt]);
    }
    __syncthreads();
  }

  for (int nt = 0; nt < 4; ++nt)
    for (int r = 0; r < 8; ++r) {
      int m = m0 + w * 16 + r + hi * 8;
      int j = n0 + nt * 16 + li;
      out[(size_t)m * DMODEL + j] = acc[nt][r];
    }
}

// =====================================================================
extern "C" void kernel_launch(void* const* d_in, const int* in_sizes, int n_in,
                              void* d_out, int out_size, void* d_ws, size_t ws_size,
                              hipStream_t stream) {
  (void)in_sizes; (void)n_in; (void)out_size; (void)ws_size;
  const float* q  = (const float*)d_in[0];
  const float* k  = (const float*)d_in[1];
  const float* v  = (const float*)d_in[2];
  const float* Wq = (const float*)d_in[3];
  const float* Wk = (const float*)d_in[4];
  const float* Wv = (const float*)d_in[5];
  const float* Wo = (const float*)d_in[6];
  float* out = (float*)d_out;

  char* ws = (char*)d_ws;
  __bf16* qh = (__bf16*)(ws);                          // 2 MB
  __bf16* kh = (__bf16*)(ws + ((size_t)2  << 20));     // 64 MB
  __bf16* vh = (__bf16*)(ws + ((size_t)66 << 20));     // 64 MB
  __bf16* ao = (__bf16*)(ws + ((size_t)130 << 20));    // 2 MB

  proj_gemm_bf16<128> <<<dim3(8,   16), 256, 0, stream>>>(q, Wq, qh);
  proj_gemm_bf16<4096><<<dim3(256, 16), 256, 0, stream>>>(k, Wk, kh);
  proj_gemm_bf16<4096><<<dim3(256, 16), 256, 0, stream>>>(v, Wv, vh);
  inv_attn            <<<dim3(128),     256, 0, stream>>>(qh, kh, vh, ao);
  out_gemm_f32        <<<dim3(8,   16), 256, 0, stream>>>(ao, Wo, out);
}
